// VQVAE_limb_hml_420906795099
// MI455X (gfx1250) — matrix operand order repaired and checked by reading, not compiled, not run on hardware
//
#include <hip/hip_runtime.h>
#include <math.h>

// ---------------------------------------------------------------------------
// VQ-VAE (T2M-GPT, 6 parts) forward on gfx1250 (wave32, WMMA).
// Every conv1d and the VQ codebook scoring run through
//   v_wmma_f32_16x16x32_f16 (f16 A/B fragments, f32 accumulate).
// Activations: f16, (B, T, C) layout, C padded to /32  -> B fragment = 16
// contiguous halves (32B load).  Weights: pre-transposed per launch to f16
// (k, Cout, CinP) -> A fragment = 2 contiguous 16B loads.  Branchless bounds
// (zero page / row clamp).  Each wave computes a 32x32 register tile
// (2x2 sub-tiles -> 4 WMMAs per K chunk); 4 waves/block tile 64x64.
// ---------------------------------------------------------------------------

typedef __attribute__((ext_vector_type(16))) _Float16 v16h;
typedef __attribute__((ext_vector_type(8)))  _Float16 v8h;
typedef __attribute__((ext_vector_type(8)))  float    v8f;

__device__ __forceinline__ v8f wmma_f16(v16h a, v16h b, v8f c) {
  return __builtin_amdgcn_wmma_f32_16x16x32_f16(false, a, false, b, (short)0, c,
                                                false, false);
}

// A (16x32 f16): lane = M row; elem e -> K = e + 16*half
// B (32x16 f16): lane%16 = N col; elem e -> K = e + 16*half
// D (16x16 f32): elem v -> M = v + 8*half (contiguous in v), N = lane%16
template <int RELU_IN>
__global__ __launch_bounds__(128) void conv1d_wmma_kernel(
    const _Float16* __restrict__ X,      // (B, Tin, CinP) f16
    const _Float16* __restrict__ Wt,     // (ks, Cout, CinP) f16
    const float*    __restrict__ Bias,   // (Cout) f32 or null
    const _Float16* __restrict__ zpage,  // >= CinP zeroed f16
    const _Float16* __restrict__ Res,    // (B, Tout, CoutP) f16 or null
    _Float16* __restrict__ Yh,           // f16 NHC out (or null)
    float*    __restrict__ Yf,           // f32 NCH out (or null)
    int CinP, int Tin, int Cout, int CoutP, int Tout,
    int ks, int st, int pad, int dil, int relu_out)
{
  const int lane  = threadIdx.x & 31;
  const int wv    = threadIdx.x >> 5;
  const int half  = lane >> 4;
  const int lm    = lane & 15;
  const int mBase = blockIdx.x * 64 + (wv >> 1) * 32;
  const int nBase = blockIdx.y * 64 + (wv & 1) * 32;   // N = 32*Tout is /64

  const int n0 = nBase + lm,      n1 = nBase + 16 + lm;
  const int b0 = n0 / Tout,       t0 = n0 % Tout;
  const int b1 = n1 / Tout,       t1 = n1 % Tout;
  int r0 = mBase + lm;        if (r0 >= Cout) r0 = Cout - 1;   // clamped rows
  int r1 = mBase + 16 + lm;   if (r1 >= Cout) r1 = Cout - 1;   // never stored

  v8f acc00 = {}, acc01 = {}, acc10 = {}, acc11 = {};
  const int hb8  = 8 * half;
  const int hb16 = 16 * half;

  for (int k = 0; k < ks; ++k) {
    const int ti0 = t0 * st - pad + k * dil;
    const int ti1 = t1 * st - pad + k * dil;
    const _Float16* __restrict__ br0 =
        (ti0 >= 0 && ti0 < Tin) ? X + (size_t)(b0 * Tin + ti0) * CinP : zpage;
    const _Float16* __restrict__ br1 =
        (ti1 >= 0 && ti1 < Tin) ? X + (size_t)(b1 * Tin + ti1) * CinP : zpage;
    const _Float16* __restrict__ ar0 = Wt + ((size_t)k * Cout + r0) * CinP;
    const _Float16* __restrict__ ar1 = Wt + ((size_t)k * Cout + r1) * CinP;

#pragma unroll 2
    for (int kc = 0; kc < CinP; kc += 32) {
      v16h bv0 = *(const v16h*)(br0 + kc + hb16);
      v16h bv1 = *(const v16h*)(br1 + kc + hb16);
      if (RELU_IN) {
        const v16h z = {};
        bv0 = __builtin_elementwise_max(bv0, z);   // v_pk_max_f16
        bv1 = __builtin_elementwise_max(bv1, z);
      }
      v8h a00 = *(const v8h*)(ar0 + kc + hb16);
      v8h a01 = *(const v8h*)(ar0 + kc + 8 + hb16);
      v8h a10 = *(const v8h*)(ar1 + kc + hb16);
      v8h a11 = *(const v8h*)(ar1 + kc + 8 + hb16);
      v16h av0 = __builtin_shufflevector(a00, a01,
                  0,1,2,3,4,5,6,7,8,9,10,11,12,13,14,15);
      v16h av1 = __builtin_shufflevector(a10, a11,
                  0,1,2,3,4,5,6,7,8,9,10,11,12,13,14,15);
      acc00 = wmma_f16(av0, bv0, acc00);
      acc01 = wmma_f16(av0, bv1, acc01);
      acc10 = wmma_f16(av1, bv0, acc10);
      acc11 = wmma_f16(av1, bv1, acc11);
    }
  }

  auto emit = [&](const v8f& acc, int mT, int bb, int tt) {
    const int cobase = mT + hb8;
    if (Yh) {
      const size_t rb = (size_t)(bb * Tout + tt) * CoutP + cobase;
      v8h pv = {};
      if (Res) pv = *(const v8h*)(Res + rb);
      v8h rv;
#pragma unroll
      for (int v = 0; v < 8; ++v) {
        float r = acc[v] + (Bias ? Bias[cobase + v] : 0.0f);
        if (relu_out) r = fmaxf(r, 0.0f);
        if (Res) r += (float)pv[v];
        rv[v] = (_Float16)r;
      }
      *(v8h*)(Yh + rb) = rv;
    } else {
#pragma unroll
      for (int v = 0; v < 8; ++v) {
        const int co = cobase + v;
        if (co >= Cout) continue;
        float r = acc[v] + (Bias ? Bias[co] : 0.0f);
        if (relu_out) r = fmaxf(r, 0.0f);
        Yf[((size_t)bb * Cout + co) * Tout + tt] = r;
      }
    }
  };
  emit(acc00, mBase,      b0, t0);
  emit(acc01, mBase,      b1, t1);
  emit(acc10, mBase + 16, b0, t0);
  emit(acc11, mBase + 16, b1, t1);
}

// ---- weight pre-transpose: f32 (Cout,Cin,ks) -> f16 (ks,Cout,CinP) ---------
__global__ void wprep_kernel(const float* __restrict__ W, _Float16* __restrict__ Wt,
                             int Cout, int Cin, int CinP, int ks) {
  const int i = blockIdx.x * 256 + threadIdx.x;
  const int tot = ks * Cout * CinP;
  if (i >= tot) return;
  const int ci = i % CinP;
  const int r  = i / CinP;
  const int co = r % Cout;
  const int k  = r / Cout;
  Wt[i] = (ci < Cin) ? (_Float16)W[((size_t)co * Cin + ci) * ks + k]
                     : (_Float16)0.0f;
}

// ---- body-part channel map (make_partseg) ----------------------------------
__device__ __forceinline__ int part_channel(int part, int j) {
  if (part == 0) {
    const int t0[7] = {0, 1, 2, 3, 193, 194, 195};
    return t0[j];
  }
  const int terms[5][5] = {{3, 6, 9, 12, 15}, {13, 16, 18, 20, 0},
                           {14, 17, 19, 21, 0}, {1, 4, 7, 10, 0},
                           {2, 5, 8, 11, 0}};
  const int nterm[5] = {5, 4, 4, 4, 4};
  const int p = part - 1, nt = nterm[p];
  if (j < nt * 12) {
    const int ti = j / 12, w = j % 12;
    const int m = terms[p][ti], i0 = m - 1;
    if (w < 3) return 4 + 3 * i0 + w;
    if (w < 9) return 67 + 6 * i0 + (w - 3);
    return 193 + 3 * m + (w - 9);
  }
  return 259 + (p - 3) * 2 + (j - nt * 12);
}

// gather x (f32 NCH) -> xc f16 (B,128,64) NHC, zero-padded channels
__global__ void gather_part_kernel(const float* __restrict__ x,
                                   _Float16* __restrict__ xc,
                                   int part, int plen) {
  const int i = blockIdx.x * 256 + threadIdx.x;
  if (i >= 32 * 128 * 64) return;
  const int j = i & 63;
  const int r = i >> 6;
  const int t = r & 127;
  const int b = r >> 7;
  float v = 0.0f;
  if (j < plen) v = x[((size_t)b * 263 + part_channel(part, j)) * 128 + t];
  xc[i] = (_Float16)v;
}

// per-sample L2 norm over 16*512 = 8192 f16 elements (f, fn both NHC f16)
__global__ __launch_bounds__(256) void l2norm_kernel(const _Float16* __restrict__ f,
                                                     _Float16* __restrict__ fn) {
  __shared__ float red[256];
  const int b = blockIdx.x, tid = threadIdx.x;
  const _Float16* p = f + (size_t)b * 8192;
  float s = 0.0f;
  for (int i = tid; i < 8192; i += 256) { float v = (float)p[i]; s += v * v; }
  red[tid] = s; __syncthreads();
  for (int o = 128; o > 0; o >>= 1) { if (tid < o) red[tid] += red[tid + o]; __syncthreads(); }
  const float inv = 1.0f / sqrtf(red[0]);
  for (int i = tid; i < 8192; i += 256)
    fn[(size_t)b * 8192 + i] = (_Float16)((float)p[i] * inv);
}

__global__ void cbnorm_kernel(const float* __restrict__ cb, float* __restrict__ cbn) {
  const int k = blockIdx.x * 128 + threadIdx.x;
  if (k >= 1024) return;
  const float* r = cb + (size_t)k * 512;
  float s = 0.0f;
  for (int c = 0; c < 512; ++c) { float v = r[c]; s += v * v; }
  cbn[k] = s;
}

// argmin over 1024 codes + straight-through gather + commit + histogram
__global__ __launch_bounds__(256) void vq_assign_kernel(
    const float* __restrict__ S,       // (B,1024,16) f32
    const float* __restrict__ cbn, const float* __restrict__ cb,
    const _Float16* __restrict__ fn,   // (B,16,512) f16
    _Float16* __restrict__ xq,         // (B,16,3072) f16
    float* __restrict__ commit, float* __restrict__ hist, int part)
{
  __shared__ float bval[256];
  __shared__ int   bidx[256];
  const int n = blockIdx.x, tid = threadIdx.x;
  const int b = n >> 4, t = n & 15;
  const float* sp = S + (size_t)b * 1024 * 16 + t;
  float best = 3.4e38f; int bi = 0;
  for (int k = tid; k < 1024; k += 256) {
    const float v = cbn[k] - 2.0f * sp[(size_t)k * 16];
    if (v < best) { best = v; bi = k; }
  }
  bval[tid] = best; bidx[tid] = bi; __syncthreads();
  for (int o = 128; o > 0; o >>= 1) {
    if (tid < o) {
      if (bval[tid + o] < bval[tid] ||
          (bval[tid + o] == bval[tid] && bidx[tid + o] < bidx[tid])) {
        bval[tid] = bval[tid + o]; bidx[tid] = bidx[tid + o];
      }
    }
    __syncthreads();
  }
  const int kk = bidx[0];
  const float* crow = cb + (size_t)kk * 512;
  float cs = 0.0f;
  for (int c = tid; c < 512; c += 256) {
    const float xd = crow[c];
    xq[((size_t)b * 16 + t) * 3072 + part * 512 + c] = (_Float16)xd;
    const float d = (float)fn[((size_t)b * 16 + t) * 512 + c] - xd;
    cs += d * d;
  }
  bval[tid] = cs; __syncthreads();
  for (int o = 128; o > 0; o >>= 1) { if (tid < o) bval[tid] += bval[tid + o]; __syncthreads(); }
  if (tid == 0) {
    atomicAdd(commit + part, bval[0]);
    atomicAdd(hist + part * 1024 + kk, 1.0f);
  }
}

// nearest x2 upsample, f16 NHC: (B,T,512) -> (B,2T,512)
__global__ void upsample2_kernel(const _Float16* __restrict__ in,
                                 _Float16* __restrict__ out, int T) {
  const int i = blockIdx.x * 256 + threadIdx.x;
  const int tot = 32 * 2 * T * 512;
  if (i >= tot) return;
  const int c = i & 511;
  const int r = i >> 9;
  const int t = r % (2 * T);
  const int b = r / (2 * T);
  out[i] = in[((size_t)b * T + (t >> 1)) * 512 + c];
}

__global__ void zerof_kernel(float* __restrict__ p, int n) {
  const int i = blockIdx.x * 256 + threadIdx.x;
  if (i < n) p[i] = 0.0f;
}
__global__ void zeroh_kernel(_Float16* __restrict__ p, int n) {
  const int i = blockIdx.x * 256 + threadIdx.x;
  if (i < n) p[i] = (_Float16)0.0f;
}

__global__ __launch_bounds__(256) void finalize_kernel(
    const float* __restrict__ commit, const float* __restrict__ hist,
    float* __restrict__ out_lp)
{
  __shared__ float red[256];
  const int tid = threadIdx.x;
  float loss = 0.0f, perp = 0.0f;
  for (int p = 0; p < 6; ++p) {
    float s = 0.0f;
    for (int k = tid; k < 1024; k += 256) {
      const float pr = hist[p * 1024 + k] * (1.0f / 512.0f);
      s += pr * logf(pr + 1e-7f);
    }
    red[tid] = s; __syncthreads();
    for (int o = 128; o > 0; o >>= 1) { if (tid < o) red[tid] += red[tid + o]; __syncthreads(); }
    perp += expf(-red[0]);
    loss += commit[p] * (1.0f / (512.0f * 512.0f));
    __syncthreads();
  }
  if (tid == 0) { out_lp[0] = loss; out_lp[1] = perp; }
}

// ---------------------------------------------------------------------------
// Host orchestration
// ---------------------------------------------------------------------------
extern "C" void kernel_launch(void* const* d_in, const int* in_sizes, int n_in,
                              void* d_out, int out_size, void* d_ws, size_t ws_size,
                              hipStream_t stream) {
  (void)in_sizes; (void)n_in; (void)out_size; (void)ws_size;

  struct CW { const float* w; const float* b; };
  int ip = 0;
  const float* x = (const float*)d_in[ip++];
  auto take = [&]() { CW c{ (const float*)d_in[ip], (const float*)d_in[ip + 1] }; ip += 2; return c; };

  CW enc_in[6], enc_out[6], enc_down[6][3], enc_res[6][3][3][2];
  for (int p = 0; p < 6; ++p) {
    enc_in[p] = take();
    for (int s = 0; s < 3; ++s) {
      enc_down[p][s] = take();
      for (int j = 0; j < 3; ++j) { enc_res[p][s][j][0] = take(); enc_res[p][s][j][1] = take(); }
    }
    enc_out[p] = take();
  }
  CW dec_in = take();
  CW dec_res[3][3][2], dec_up[3];
  for (int s = 0; s < 3; ++s) {
    for (int j = 0; j < 3; ++j) { dec_res[s][j][0] = take(); dec_res[s][j][1] = take(); }
    dec_up[s] = take();
  }
  CW dec_mid = take(), dec_out = take();
  const float* cbs[6];
  for (int p = 0; p < 6; ++p) cbs[p] = (const float*)d_in[ip++];

  // ---- workspace carve-up ----
  float* wsf = (float*)d_ws; size_t of = 0;
  float* S      = wsf + of; of += 524288;           // 32*1024*16
  float* cbn    = wsf + of; of += 1024;
  float* commit = wsf + of; of += 8;
  float* hist   = wsf + of; of += 6144;             // commit+hist contiguous
  _Float16* wsh = (_Float16*)(wsf + of);
  size_t oh = 0;
  _Float16* zpage = wsh + oh; oh += 4096;
  _Float16* xc    = wsh + oh; oh += 262144;         // 32*128*64
  _Float16* fbuf  = wsh + oh; oh += 262144;         // 32*16*512
  _Float16* fn    = wsh + oh; oh += 262144;
  _Float16* xq    = wsh + oh; oh += 1572864;        // 32*16*3072
  _Float16* hA    = wsh + oh; oh += 2097152;        // 32*128*512
  _Float16* hB    = wsh + oh; oh += 2097152;
  _Float16* hR    = wsh + oh; oh += 2097152;
  _Float16* wcur  = wsh + oh;                       // f16 weight arena
  float* out = (float*)d_out;   // [0..1077247]=xd, [1077248]=loss, [1077249]=perp

  struct PW { const _Float16* w; const float* b; };
  auto prep = [&](CW c, int Cout, int Cin, int CinP, int ks) -> PW {
    _Float16* dst = wcur;
    const size_t tot = (size_t)ks * Cout * CinP;
    wcur += tot;
    wprep_kernel<<<((int)tot + 255) / 256, 256, 0, stream>>>(c.w, dst, Cout, Cin, CinP, ks);
    return PW{dst, c.b};
  };

  auto conv = [&](const _Float16* X, PW w, const _Float16* Rs,
                  _Float16* Yh, float* Yf,
                  int CinP, int Tin, int Cout, int CoutP, int Tout,
                  int ks, int st, int pad, int dil, int ri, int ro) {
    const int N = 32 * Tout;
    dim3 grid((Cout + 63) / 64, (N + 63) / 64);
    if (ri)
      conv1d_wmma_kernel<1><<<grid, 128, 0, stream>>>(X, w.w, w.b, zpage, Rs, Yh, Yf,
                                                      CinP, Tin, Cout, CoutP, Tout,
                                                      ks, st, pad, dil, ro);
    else
      conv1d_wmma_kernel<0><<<grid, 128, 0, stream>>>(X, w.w, w.b, zpage, Rs, Yh, Yf,
                                                      CinP, Tin, Cout, CoutP, Tout,
                                                      ks, st, pad, dil, ro);
  };

  zerof_kernel<<<(6152 + 255) / 256, 256, 0, stream>>>(commit, 6152);
  zeroh_kernel<<<16, 256, 0, stream>>>(zpage, 4096);

  const int plen[6] = {7, 60, 48, 48, 50, 50};

  // ---- per-part encoder + VQ ----
  for (int p = 0; p < 6; ++p) {
    gather_part_kernel<<<(32 * 128 * 64 + 255) / 256, 256, 0, stream>>>(x, xc, p, plen[p]);

    PW w_in = prep(enc_in[p], 512, plen[p], 64, 3);
    conv(xc, w_in, nullptr, hA, nullptr, 64, 128, 512, 512, 128, 3, 1, 1, 1, 0, 1);

    _Float16* cur = hA; _Float16* alt = hB;
    int T = 128;
    for (int s = 0; s < 3; ++s) {
      PW w_dn = prep(enc_down[p][s], 512, 512, 512, 4);
      conv(cur, w_dn, nullptr, alt, nullptr, 512, T, 512, 512, T / 2, 4, 2, 1, 1, 0, 0);
      T /= 2; { _Float16* t2 = cur; cur = alt; alt = t2; }
      int d = 1;
      for (int j = 0; j < 3; ++j) {
        PW c1 = prep(enc_res[p][s][j][0], 512, 512, 512, 3);
        PW c2 = prep(enc_res[p][s][j][1], 512, 512, 512, 1);
        conv(cur, c1, nullptr, hR, nullptr, 512, T, 512, 512, T, 3, 1, d, d, 1, 0);
        conv(hR,  c2, cur,     cur, nullptr, 512, T, 512, 512, T, 1, 1, 0, 1, 1, 0);
        d *= 3;
      }
    }
    PW w_out = prep(enc_out[p], 512, 512, 512, 3);
    conv(cur, w_out, nullptr, fbuf, nullptr, 512, 16, 512, 512, 16, 3, 1, 1, 1, 0, 0);
    l2norm_kernel<<<32, 256, 0, stream>>>(fbuf, fn);

    cbnorm_kernel<<<8, 128, 0, stream>>>(cbs[p], cbn);
    PW pcb = prep(CW{cbs[p], nullptr}, 1024, 512, 512, 1);
    conv(fn, pcb, nullptr, nullptr, S, 512, 16, 1024, 1024, 16, 1, 1, 0, 1, 0, 0);
    vq_assign_kernel<<<512, 256, 0, stream>>>(S, cbn, cbs[p], fn, xq, commit, hist, p);
  }

  // ---- decoder ----
  PW w_di = prep(dec_in, 512, 3072, 3072, 3);
  conv(xq, w_di, nullptr, hA, nullptr, 3072, 16, 512, 512, 16, 3, 1, 1, 1, 0, 1);
  int T = 16;
  for (int s = 0; s < 3; ++s) {
    int d = 9;
    for (int j = 0; j < 3; ++j) {
      PW c1 = prep(dec_res[s][j][0], 512, 512, 512, 3);
      PW c2 = prep(dec_res[s][j][1], 512, 512, 512, 1);
      conv(hA, c1, nullptr, hR, nullptr, 512, T, 512, 512, T, 3, 1, d, d, 1, 0);
      conv(hR, c2, hA,      hA, nullptr, 512, T, 512, 512, T, 1, 1, 0, 1, 1, 0);
      d /= 3;
    }
    upsample2_kernel<<<(32 * 512 * 2 * T + 255) / 256, 256, 0, stream>>>(hA, hB, T);
    T *= 2;
    PW w_up = prep(dec_up[s], 512, 512, 512, 3);
    conv(hB, w_up, nullptr, hA, nullptr, 512, T, 512, 512, T, 3, 1, 1, 1, 0, 0);
  }
  PW w_md = prep(dec_mid, 512, 512, 512, 3);
  conv(hA, w_md, nullptr, hB, nullptr, 512, 128, 512, 512, 128, 3, 1, 1, 1, 0, 1);
  PW w_ot = prep(dec_out, 263, 512, 512, 3);
  conv(hB, w_ot, nullptr, nullptr, out, 512, 128, 263, 512, 128, 3, 1, 1, 1, 0, 0);

  finalize_kernel<<<1, 256, 0, stream>>>(commit, hist, out + 1077248);
}
